// Decoder_3023656976622
// MI455X (gfx1250) — matrix operand order repaired and checked by reading, not compiled, not run on hardware
//
#include <hip/hip_runtime.h>
#include <hip/hip_bf16.h>

// ---------------- model constants ----------------
#define BATCH 32
#define TIN   512
#define TOUT  512
#define ENC   512
#define NOUT  80
#define HID   1024
#define PRE   256
#define ATT   128
#define CCH   32
#define KW    31
#define G4    (4*HID)          // 4096 gates
#define KD0   (ENC+PRE+HID)    // 1792 : lstm0 [lx | h0]
#define KD1   (HID+HID)        // 2048 : lstm1 [h0 | h1]
#define KD_A0 (ENC+PRE)        // 768

typedef __bf16 bf16;
typedef __attribute__((ext_vector_type(16))) __bf16 v16bf;
typedef __attribute__((ext_vector_type(8)))  __bf16 v8bf;
typedef __attribute__((ext_vector_type(8)))  float   v8f;

// ---------------- helpers ----------------
__device__ __forceinline__ float sigf(float x) { return 1.0f / (1.0f + __expf(-x)); }

__device__ __forceinline__ unsigned mixhash(unsigned a, unsigned b, unsigned c) {
    unsigned x = a * 1664525u ^ b * 22695477u ^ c * 2654435761u;
    x ^= x >> 13; x *= 2246822519u; x ^= x >> 16;
    return x;
}

// ---------------- prep kernels (once per launch) ----------------
__global__ __launch_bounds__(256) void prep_wcat0(const float* __restrict__ Wih,
                                                  const float* __restrict__ Whh,
                                                  bf16* __restrict__ out) {
    size_t i = (size_t)blockIdx.x * 256 + threadIdx.x;
    if (i >= (size_t)G4 * KD0) return;
    int n = (int)(i / KD0), col = (int)(i % KD0);
    float v = (col < (ENC + PRE)) ? Wih[(size_t)n * (ENC + PRE) + col]
                                  : Whh[(size_t)n * HID + (col - (ENC + PRE))];
    out[i] = (bf16)v;
}

__global__ __launch_bounds__(256) void prep_wcat1(const float* __restrict__ Wih,
                                                  const float* __restrict__ Whh,
                                                  bf16* __restrict__ out) {
    size_t i = (size_t)blockIdx.x * 256 + threadIdx.x;
    if (i >= (size_t)G4 * KD1) return;
    int n = (int)(i / KD1), col = (int)(i % KD1);
    float v = (col < HID) ? Wih[(size_t)n * HID + col]
                          : Whh[(size_t)n * HID + (col - HID)];
    out[i] = (bf16)v;
}

__global__ __launch_bounds__(256) void f32_to_bf16(const float* __restrict__ in,
                                                   bf16* __restrict__ out, size_t n) {
    size_t i = (size_t)blockIdx.x * 256 + threadIdx.x;
    if (i < n) out[i] = (bf16)in[i];
}

// att_enc_W [ENC,ATT] -> Wt [ATT,ENC] bf16 (row-major [N,K] for GEMM)
__global__ __launch_bounds__(256) void prep_encW_t(const float* __restrict__ W,
                                                   bf16* __restrict__ out) {
    int i = blockIdx.x * 256 + threadIdx.x;
    if (i >= ATT * ENC) return;
    int a = i / ENC, e = i % ENC;
    out[i] = (bf16)W[(size_t)e * ATT + a];
}

__global__ __launch_bounds__(256) void zero_f32(float* p, size_t n) {
    size_t i = (size_t)blockIdx.x * 256 + threadIdx.x;
    if (i < n) p[i] = 0.0f;
}
__global__ __launch_bounds__(256) void zero_bf16(bf16* p, size_t n) {
    size_t i = (size_t)blockIdx.x * 256 + threadIdx.x;
    if (i < n) p[i] = (bf16)0.0f;
}
__global__ __launch_bounds__(64) void zero_u32(unsigned* p, int n) {
    int i = blockIdx.x * 64 + threadIdx.x;
    if (i < n) p[i] = 0u;
}

// ---------------- prenet for ALL steps (teacher forcing => step-independent) ----------------
__global__ __launch_bounds__(256) void prenet_all(const float* __restrict__ targets, // [B,TOUT,NOUT]
                                                  const float* __restrict__ W0,      // [NOUT,PRE]
                                                  const float* __restrict__ b0,
                                                  const float* __restrict__ W1,      // [PRE,PRE]
                                                  const float* __restrict__ b1,
                                                  bf16* __restrict__ xall) {         // [TOUT*B, PRE]
    int row = blockIdx.x;                 // t*B + b
    int t = row / BATCH, b = row % BATCH;
    int tid = threadIdx.x;
    __shared__ float in0[NOUT];
    __shared__ float x0[PRE];
    if (tid < NOUT)
        in0[tid] = (t == 0) ? 0.0f : targets[((size_t)b * TOUT + (t - 1)) * NOUT + tid];
    __syncthreads();
    {   // layer 0 + relu + dropout(0.5)
        float s = b0[tid];
        for (int i = 0; i < NOUT; ++i) s += in0[i] * W0[(size_t)i * PRE + tid];
        s = fmaxf(s, 0.0f);
        unsigned u = mixhash((unsigned)(2 * t), (unsigned)b, (unsigned)tid);
        x0[tid] = (u & 1u) ? s * 2.0f : 0.0f;
    }
    __syncthreads();
    {   // layer 1 + relu + dropout(0.5)
        float s = b1[tid];
        for (int i = 0; i < PRE; ++i) s += x0[i] * W1[(size_t)i * PRE + tid];
        s = fmaxf(s, 0.0f);
        unsigned u = mixhash((unsigned)(2 * t + 1), (unsigned)b, (unsigned)tid);
        float xv = (u & 1u) ? s * 2.0f : 0.0f;
        xall[(size_t)row * PRE + tid] = (bf16)xv;
    }
}

// ---------------- generic WMMA GEMM + (optional) grid-fused LSTM cell ----------------
// C[M,N] = A[M,Kd] * W[N,Kd]^T + bias   (A,W bf16 row-major; C f32)
// One wave per 16x16 tile via v_wmma_f32_16x16x32_bf16.
// If do_cell: after an in-kernel grid barrier (atomic arrive + generation flag + s_sleep
// spin), the LSTM cell + zoneout over gates C is computed, distributed across the grid,
// writing h/c f32 state and bf16 h into the next GEMM's A operand slice.
__global__ __launch_bounds__(256) void wmma_gemm_bf16(const bf16* __restrict__ A, int lda,
                                                      const bf16* __restrict__ W,
                                                      const float* __restrict__ bias,
                                                      float* __restrict__ C, int ldc,
                                                      int Mtiles, int Ntiles, int Kd,
                                                      float* h, float* c,
                                                      bf16* Abuf, int aoff, int ald,
                                                      unsigned* bar_cnt, unsigned* bar_gen,
                                                      int do_cell) {
    int wave = blockIdx.x * (blockDim.x >> 5) + (threadIdx.x >> 5);
    int lane = threadIdx.x & 31;
    if (wave < Mtiles * Ntiles) {
        int mt = wave / Ntiles, nt = wave % Ntiles;
        int hsel = lane >> 4;          // 0: lanes 0-15, 1: lanes 16-31
        int l16  = lane & 15;
        const bf16* Arow = A + (size_t)(mt * 16 + l16) * lda;   // A fragment row M=l16
        const bf16* Wrow = W + (size_t)(nt * 16 + l16) * Kd;    // B fragment col N=l16
        v8f acc = {};
        for (int k = 0; k < Kd; k += 32) {
            // A 16x32 bf16 layout: lane<16 holds K {k..k+7, k+16..k+23}; lane>=16 shifted by 8
            int ka = k + hsel * 8;
            v16bf a;
            *(v8bf*)&a         = *(const v8bf*)(Arow + ka);
            *(((v8bf*)&a) + 1) = *(const v8bf*)(Arow + ka + 16);
            // B 32x16 bf16 layout: lane<16 holds K {k..k+7, k+16..k+23}; lane>=16 shifted by 8
            int kb = k + hsel * 8;
            v16bf bfrag;
            *(v8bf*)&bfrag         = *(const v8bf*)(Wrow + kb);
            *(((v8bf*)&bfrag) + 1) = *(const v8bf*)(Wrow + kb + 16);
            __builtin_prefetch(Wrow + kb + 128, 0, 3);   // global_prefetch_b8 toward near cache
            acc = __builtin_amdgcn_wmma_f32_16x16x32_bf16(false, a, false, bfrag,
                                                          (short)0, acc, false, false);
        }
        int ncol = nt * 16 + l16;
        float bv = bias ? bias[ncol] : 0.0f;
        float* Cp = C + (size_t)(mt * 16 + hsel * 8) * ldc + ncol;
#pragma unroll
        for (int r = 0; r < 8; ++r) Cp[(size_t)r * ldc] = acc[r] + bv;
    }
    if (!do_cell) return;

    // ---- grid barrier: all gate tiles stored -> visible to all blocks ----
    __syncthreads();
    if (threadIdx.x == 0) {
        __threadfence();                                    // release gate stores
        unsigned gv = *(volatile unsigned*)bar_gen;
        unsigned arrived = atomicAdd(bar_cnt, 1u) + 1u;
        if (arrived == (unsigned)gridDim.x) {
            *(volatile unsigned*)bar_cnt = 0u;              // reset for next use
            __threadfence();
            atomicAdd(bar_gen, 1u);                         // open the gate
        } else {
            while (*(volatile unsigned*)bar_gen == gv) __builtin_amdgcn_s_sleep(1);
        }
        __threadfence();                                    // acquire
    }
    __syncthreads();

    // ---- distributed LSTM cell + zoneout over [BATCH,HID] ----
    int stride = (int)gridDim.x * 256;
    for (int idx = blockIdx.x * 256 + threadIdx.x; idx < BATCH * HID; idx += stride) {
        int b = idx >> 10, j = idx & (HID - 1);
        const float* gp = C + (size_t)b * G4;
        float gi = gp[j], gf = gp[j + HID], gg = gp[j + 2 * HID], go = gp[j + 3 * HID];
        float cn = sigf(gf) * c[idx] + sigf(gi) * tanhf(gg);
        float hn = sigf(go) * tanhf(cn);
        float hz = 0.1f * h[idx] + 0.9f * hn;               // zoneout blend
        float cz = 0.1f * c[idx] + 0.9f * cn;
        h[idx] = hz; c[idx] = cz;
        Abuf[(size_t)b * ald + aoff + j] = (bf16)hz;
    }
}

// ---------------- attention (one block per batch row) + fused projection of step t-1 ----------------
__global__ __launch_bounds__(256) void attention_kernel(const float* __restrict__ enc,      // [B,TIN,ENC]
                                                        const float* __restrict__ enc_proj, // [B,TIN,ATT]
                                                        const int*   __restrict__ in_lens,
                                                        const float* __restrict__ h0,       // [B,HID]
                                                        const float* __restrict__ Wdec,     // [HID,ATT]
                                                        const float* __restrict__ convW,    // [CCH,1,KW]
                                                        const float* __restrict__ locW,     // [CCH,ATT]
                                                        const float* __restrict__ vvec,     // [ATT]
                                                        const float* __restrict__ vb,       // scalar
                                                        float* __restrict__ acc_att,        // [B,TIN]
                                                        float* __restrict__ attws_out,      // [B,TOUT,TIN]
                                                        float* __restrict__ att_c,          // [B,ENC]
                                                        bf16*  __restrict__ A0,             // [B,KD_A0]
                                                        const bf16* __restrict__ xall,      // [TOUT*B,PRE]
                                                        const float* __restrict__ h1,       // [B,HID]
                                                        const float* __restrict__ featW,    // [HID+ENC,NOUT]
                                                        const float* __restrict__ probW,    // [HID+ENC,1]
                                                        const float* __restrict__ probB,
                                                        float* __restrict__ outs,           // [B,NOUT,TOUT]
                                                        float* __restrict__ logits,         // [B,TOUT]
                                                        int t) {
    __shared__ float sm_att[TIN + KW - 1];
    __shared__ float sm_dec[ATT];
    __shared__ float sm_e[TIN];
    __shared__ float red[256];
    __shared__ float sm_hcs[HID + ENC];
    const int b = blockIdx.x, tid = threadIdx.x;
    const int len = in_lens[b];

    // ---- fused projection of previous step (reads h1/att_c before this step mutates them) ----
    if (t > 0) {
        for (int j = tid; j < HID; j += 256) sm_hcs[j] = h1[(size_t)b * HID + j];
        for (int j = tid; j < ENC; j += 256) sm_hcs[HID + j] = att_c[(size_t)b * ENC + j];
        __syncthreads();
        if (tid < NOUT) {
            float s = 0.0f;
            for (int j = 0; j < HID + ENC; ++j) s += sm_hcs[j] * featW[(size_t)j * NOUT + tid];
            outs[((size_t)b * NOUT + tid) * TOUT + (t - 1)] = s;
        } else if (tid == NOUT) {
            float s = probB[0];
            for (int j = 0; j < HID + ENC; ++j) s += sm_hcs[j] * probW[j];
            logits[(size_t)b * TOUT + (t - 1)] = s;
        }
    }

    for (int i = tid; i < TIN + KW - 1; i += 256) sm_att[i] = 0.0f;
    __syncthreads();
    for (int s = tid; s < TIN; s += 256) {
        float v = (t == 0) ? ((s < len) ? 1.0f / (float)len : 0.0f)
                           : acc_att[(size_t)b * TIN + s];
        sm_att[s + (KW / 2)] = v;
    }
    if (tid < ATT) {                       // dec = h0 @ att_dec_W
        float d = 0.0f;
        const float* hp = h0 + (size_t)b * HID;
        for (int k = 0; k < HID; ++k) d += hp[k] * Wdec[(size_t)k * ATT + tid];
        sm_dec[tid] = d;
    }
    __syncthreads();

    for (int s = tid; s < TIN; s += 256) {     // conv -> loc -> tanh energy
        float convc[CCH];
#pragma unroll
        for (int c = 0; c < CCH; ++c) convc[c] = 0.0f;
        for (int kk = 0; kk < KW; ++kk) {
            float av = sm_att[s + kk];
#pragma unroll
            for (int c = 0; c < CCH; ++c) convc[c] += av * convW[c * KW + kk];
        }
        const float* ep = enc_proj + ((size_t)b * TIN + s) * ATT;
        float e = 0.0f;
        for (int a = 0; a < ATT; ++a) {
            float loc = 0.0f;
#pragma unroll
            for (int c = 0; c < CCH; ++c) loc += convc[c] * locW[(size_t)c * ATT + a];
            e += vvec[a] * tanhf(ep[a] + sm_dec[a] + loc);
        }
        e += vb[0];
        if (s >= len) e = -1e9f;
        sm_e[s] = e;
    }
    __syncthreads();

    // softmax over TIN
    float lmax = -1e30f;
    for (int s = tid; s < TIN; s += 256) lmax = fmaxf(lmax, sm_e[s]);
    red[tid] = lmax; __syncthreads();
    for (int off = 128; off > 0; off >>= 1) {
        if (tid < off) red[tid] = fmaxf(red[tid], red[tid + off]);
        __syncthreads();
    }
    float m = red[0]; __syncthreads();
    float lsum = 0.0f;
    for (int s = tid; s < TIN; s += 256) { float w = __expf(sm_e[s] - m); sm_e[s] = w; lsum += w; }
    red[tid] = lsum; __syncthreads();
    for (int off = 128; off > 0; off >>= 1) {
        if (tid < off) red[tid] += red[tid + off];
        __syncthreads();
    }
    float inv = 1.0f / red[0]; __syncthreads();
    for (int s = tid; s < TIN; s += 256) {
        float w = sm_e[s] * inv;
        sm_e[s] = w;
        attws_out[((size_t)b * TOUT + t) * TIN + s] = w;
        acc_att[(size_t)b * TIN + s] = ((t == 0) ? 0.0f : acc_att[(size_t)b * TIN + s]) + w;
    }
    __syncthreads();

    // context att_c[b] = att_w @ enc[b]
    for (int e = tid; e < ENC; e += 256) {
        float s = 0.0f;
        const float* ep = enc + (size_t)b * TIN * ENC + e;
        for (int si = 0; si < TIN; ++si) s += sm_e[si] * ep[(size_t)si * ENC];
        att_c[(size_t)b * ENC + e] = s;
        A0[(size_t)b * KD_A0 + e] = (bf16)s;           // GEMM A operand (bf16)
    }
    for (int j = tid; j < PRE; j += 256)                // append prenet x_t
        A0[(size_t)b * KD_A0 + ENC + j] = xall[((size_t)t * BATCH + b) * PRE + j];
}

// ---------------- final-step projection ----------------
__global__ __launch_bounds__(128) void proj_kernel(const float* __restrict__ h1,
                                                   const float* __restrict__ att_c,
                                                   const float* __restrict__ featW,
                                                   const float* __restrict__ probW,
                                                   const float* __restrict__ probB,
                                                   float* __restrict__ outs,
                                                   float* __restrict__ logits,
                                                   int t) {
    __shared__ float hcs[HID + ENC];
    int b = blockIdx.x, tid = threadIdx.x;
    for (int j = tid; j < HID; j += 128) hcs[j] = h1[(size_t)b * HID + j];
    for (int j = tid; j < ENC; j += 128) hcs[HID + j] = att_c[(size_t)b * ENC + j];
    __syncthreads();
    if (tid < NOUT) {
        float s = 0.0f;
        for (int j = 0; j < HID + ENC; ++j) s += hcs[j] * featW[(size_t)j * NOUT + tid];
        outs[((size_t)b * NOUT + tid) * TOUT + t] = s;
    } else if (tid == NOUT) {
        float s = probB[0];
        for (int j = 0; j < HID + ENC; ++j) s += hcs[j] * probW[j];
        logits[(size_t)b * TOUT + t] = s;
    }
}

// ---------------- host side ----------------
static inline char* carve(char*& p, size_t bytes) {
    char* r = p;
    p += (bytes + 255) & ~(size_t)255;
    return r;
}

extern "C" void kernel_launch(void* const* d_in, const int* in_sizes, int n_in,
                              void* d_out, int out_size, void* d_ws, size_t ws_size,
                              hipStream_t stream) {
    (void)in_sizes; (void)n_in; (void)out_size; (void)ws_size;
    const float* enc        = (const float*)d_in[0];
    const int*   in_lens    = (const int*)  d_in[1];
    const float* targets    = (const float*)d_in[2];
    const float* prenet_W0  = (const float*)d_in[3];
    const float* prenet_b0  = (const float*)d_in[4];
    const float* prenet_W1  = (const float*)d_in[5];
    const float* prenet_b1  = (const float*)d_in[6];
    const float* lstm0_Wih  = (const float*)d_in[7];
    const float* lstm0_Whh  = (const float*)d_in[8];
    const float* lstm0_b    = (const float*)d_in[9];
    const float* lstm1_Wih  = (const float*)d_in[10];
    const float* lstm1_Whh  = (const float*)d_in[11];
    const float* lstm1_b    = (const float*)d_in[12];
    const float* att_enc_W  = (const float*)d_in[13];
    const float* att_enc_b  = (const float*)d_in[14];
    const float* att_dec_W  = (const float*)d_in[15];
    const float* att_loc_W  = (const float*)d_in[16];
    const float* att_conv   = (const float*)d_in[17];
    const float* att_v      = (const float*)d_in[18];
    const float* att_v_b    = (const float*)d_in[19];
    const float* feat_W     = (const float*)d_in[20];
    const float* prob_W     = (const float*)d_in[21];
    const float* prob_b     = (const float*)d_in[22];

    float* outs   = (float*)d_out;                                    // [B,NOUT,TOUT]
    float* logits = outs + (size_t)BATCH * NOUT * TOUT;               // [B,TOUT]
    float* attws  = logits + (size_t)BATCH * TOUT;                    // [B,TOUT,TIN]

    // workspace carve (bf16 weights sized to stay resident in the 192MB L2)
    char* p = (char*)d_ws;
    bf16*  wcat0    = (bf16*) carve(p, (size_t)G4 * KD0 * sizeof(bf16));   // 14.7 MB
    bf16*  wcat1    = (bf16*) carve(p, (size_t)G4 * KD1 * sizeof(bf16));   // 16.8 MB
    bf16*  enc_bf   = (bf16*) carve(p, (size_t)BATCH * TIN * ENC * sizeof(bf16)); // 16.8 MB
    bf16*  encWt    = (bf16*) carve(p, (size_t)ATT * ENC * sizeof(bf16));
    float* enc_proj = (float*)carve(p, (size_t)BATCH * TIN * ATT * sizeof(float)); // 8.4 MB
    bf16*  xall     = (bf16*) carve(p, (size_t)TOUT * BATCH * PRE * sizeof(bf16)); // 8.4 MB
    bf16*  A0       = (bf16*) carve(p, (size_t)BATCH * KD_A0 * sizeof(bf16));
    bf16*  A1       = (bf16*) carve(p, (size_t)BATCH * KD1 * sizeof(bf16));
    float* gates    = (float*)carve(p, (size_t)BATCH * G4 * sizeof(float));
    float* h0       = (float*)carve(p, (size_t)BATCH * HID * sizeof(float));
    float* c0       = (float*)carve(p, (size_t)BATCH * HID * sizeof(float));
    float* h1       = (float*)carve(p, (size_t)BATCH * HID * sizeof(float));
    float* c1       = (float*)carve(p, (size_t)BATCH * HID * sizeof(float));
    float* acc_att  = (float*)carve(p, (size_t)BATCH * TIN * sizeof(float));
    float* att_c    = (float*)carve(p, (size_t)BATCH * ENC * sizeof(float));
    unsigned* bar   = (unsigned*)carve(p, 2 * sizeof(unsigned));      // [cnt, gen]

    // ---- one-time prep (per launch; deterministic) ----
    {
        size_t n0 = (size_t)G4 * KD0;
        prep_wcat0<<<(unsigned)((n0 + 255) / 256), 256, 0, stream>>>(lstm0_Wih, lstm0_Whh, wcat0);
        size_t n1 = (size_t)G4 * KD1;
        prep_wcat1<<<(unsigned)((n1 + 255) / 256), 256, 0, stream>>>(lstm1_Wih, lstm1_Whh, wcat1);
        size_t ne = (size_t)BATCH * TIN * ENC;
        f32_to_bf16<<<(unsigned)((ne + 255) / 256), 256, 0, stream>>>(enc, enc_bf, ne);
        prep_encW_t<<<(ATT * ENC + 255) / 256, 256, 0, stream>>>(att_enc_W, encWt);
        zero_f32<<<((4 * BATCH * HID + BATCH * TIN) + 255) / 256, 256, 0, stream>>>(
            h0, (size_t)4 * BATCH * HID + (size_t)BATCH * TIN);
        zero_bf16<<<((BATCH * KD1) + 255) / 256, 256, 0, stream>>>(A1, (size_t)BATCH * KD1);
        zero_u32<<<1, 64, 0, stream>>>(bar, 2);
        // enc_proj = enc @ att_enc_W + b : WMMA GEMM, M=16384, N=128, K=512 (no cell)
        wmma_gemm_bf16<<<(16384 / 16) * (ATT / 16) / 8, 256, 0, stream>>>(
            enc_bf, ENC, encWt, att_enc_b, enc_proj, ATT, 16384 / 16, ATT / 16, ENC,
            nullptr, nullptr, nullptr, 0, 0, nullptr, nullptr, 0);
        // prenet for all steps (teacher forcing => hoisted out of recurrence)
        prenet_all<<<TOUT * BATCH, 256, 0, stream>>>(targets, prenet_W0, prenet_b0,
                                                     prenet_W1, prenet_b1, xall);
    }

    // ---- sequential decode loop: 3 stream-ordered kernels per step ----
    const int blk = ((BATCH / 16) * (G4 / 16)) / 8;   // 512 tiles / 8 waves = 64 blocks
    for (int t = 0; t < TOUT; ++t) {
        // attention for t (+ fused projection of t-1)
        attention_kernel<<<BATCH, 256, 0, stream>>>(enc, enc_proj, in_lens, h0,
                                                    att_dec_W, att_conv, att_loc_W,
                                                    att_v, att_v_b, acc_att, attws,
                                                    att_c, A0, xall,
                                                    h1, feat_W, prob_W, prob_b,
                                                    outs, logits, t);
        // gates0 = [att_c|x] @ [Wih|Whh]^T + b, then grid-fused cell0 -> h0,c0, A1[:, :HID]
        wmma_gemm_bf16<<<blk, 256, 0, stream>>>(A0, KD_A0, wcat0, lstm0_b,
                                                gates, G4, BATCH / 16, G4 / 16, KD0,
                                                h0, c0, A1, 0, KD1, bar, bar + 1, 1);
        // gates1 = [h0|h1] @ [Wih|Whh]^T + b, then grid-fused cell1 -> h1,c1, A1[:, HID:]
        wmma_gemm_bf16<<<blk, 256, 0, stream>>>(A1, KD1, wcat1, lstm1_b,
                                                gates, G4, BATCH / 16, G4 / 16, KD1,
                                                h1, c1, A1, HID, KD1, bar, bar + 1, 1);
    }
    // projection for the last step
    proj_kernel<<<BATCH, 128, 0, stream>>>(h1, att_c, feat_W, prob_W, prob_b,
                                           outs, logits, TOUT - 1);
}
